// HeteroGATv2_13537736917229
// MI455X (gfx1250) — matrix operand order repaired and checked by reading, not compiled, not run on hardware
//
#include <hip/hip_runtime.h>
#include <hip/hip_bf16.h>
#include <math.h>

// ---------------------------------------------------------------------------
// HeteroGATv2 forward for gfx1250 (MI455X).  GEMMs run on bf16 WMMA
// (v_wmma_f32_16x16x32_bf16); edge softmax/aggregation are wave32 gather /
// L2-resident atomic-scatter kernels.
// ---------------------------------------------------------------------------

typedef __attribute__((ext_vector_type(16))) __bf16 v16bf;
typedef __attribute__((ext_vector_type(8)))  __bf16 v8bf;
typedef __attribute__((ext_vector_type(8)))  float  v8f;

#define HID 128
#define HC  512   // H * C
#define NH  4

// ----------------------------- device helpers ------------------------------

static __device__ __forceinline__ unsigned short f2bf(float f) {
  unsigned u = __float_as_uint(f);
  u += 0x7FFFu + ((u >> 16) & 1u);          // round-to-nearest-even
  return (unsigned short)(u >> 16);
}

// monotone float -> uint key (order preserving) for atomicMax-based segment max
static __device__ __forceinline__ unsigned fkey(float f) {
  unsigned b = __float_as_uint(f);
  return (b & 0x80000000u) ? ~b : (b | 0x80000000u);
}
static __device__ __forceinline__ float keyf(unsigned k) {
  unsigned b = (k & 0x80000000u) ? (k & 0x7FFFFFFFu) : ~k;
  return __uint_as_float(b);
}

// A fragment (16x32 bf16, MxK): lane L holds row M=L%16, group g=L/16.
// elem i<8  -> K = k0 + g*8 + i ; elem i>=8 -> K = k0 + 16 + g*8 + (i-8)
static __device__ __forceinline__ v16bf load_a_frag(const __bf16* __restrict__ row,
                                                    int k0, int g) {
  v8bf lo = *(const v8bf*)(row + k0 + g * 8);
  v8bf hi = *(const v8bf*)(row + k0 + 16 + g * 8);
  v16bf f;
#pragma unroll
  for (int i = 0; i < 8; ++i) { f[i] = lo[i]; f[i + 8] = hi[i]; }
  return f;
}

// B fragment (32x16 bf16, KxN): lane L holds col N=L%16, group g=L/16.
// elem i<8  -> K = k0 + g*8 + i ; elem i>=8 -> K = k0 + 16 + g*8 + (i-8)
static __device__ __forceinline__ v16bf load_b_frag(const __bf16* __restrict__ col,
                                                    int k0, int g) {
  v8bf lo = *(const v8bf*)(col + k0 + g * 8);
  v8bf hi = *(const v8bf*)(col + k0 + 16 + g * 8);
  v16bf f;
#pragma unroll
  for (int i = 0; i < 8; ++i) { f[i] = lo[i]; f[i + 8] = hi[i]; }
  return f;
}

// ------------------------------- tiny kernels ------------------------------

__global__ void k_tobf(const float* __restrict__ x, unsigned short* __restrict__ y,
                       long n) {
  long i = (long)blockIdx.x * blockDim.x + threadIdx.x;
  if (i < n) y[i] = f2bf(x[i]);
}

// W [K, Ncol] f32 (row-major)  ->  Wt [Ncol, K] bf16 (K contiguous per column)
__global__ void k_wtrans(const float* __restrict__ W, unsigned short* __restrict__ Wt,
                         int K, int Ncol) {
  int i = blockIdx.x * blockDim.x + threadIdx.x;
  if (i >= K * Ncol) return;
  int k = i / Ncol, n = i - k * Ncol;
  Wt[(size_t)n * K + k] = f2bf(W[i]);
}

__global__ void k_zero(unsigned* __restrict__ p, long n) {
  long i = (long)blockIdx.x * blockDim.x + threadIdx.x;
  if (i < n) p[i] = 0u;
}

// hsum[n,c] = bias_rel0[c] + bias_rel1[c]
__global__ void k_init_hsum(float* __restrict__ hsum, const float* __restrict__ ba,
                            const float* __restrict__ bb, long n) {
  long i = (long)blockIdx.x * blockDim.x + threadIdx.x;
  if (i < n) {
    int c = (int)(i & (HID - 1));
    hsum[i] = ba[c] + bb[c];
  }
}

// ----------------------------- WMMA bf16 GEMM ------------------------------
// C[M,Ncol] = A[M,K](bf16) x B[K,Ncol] + bias ; Bt stored [Ncol,K] bf16.
// One wave -> 16x64 output tile (4 accumulators).  K%32==0, Ncol%64==0.

__global__ void k_gemm_bf16(const unsigned short* __restrict__ A,
                            const unsigned short* __restrict__ Bt,
                            const float* __restrict__ bias,
                            float* __restrict__ C,
                            int M, int K, int Ncol) {
  int lane = threadIdx.x & 31;
  int wave = threadIdx.x >> 5;
  int m0 = (blockIdx.x * 8 + wave) * 16;
  int n0 = blockIdx.y * 64;
  int rlo = lane & 15;
  int g = lane >> 4;

  int row = m0 + rlo;
  int rowc = (row < M) ? row : (M - 1);       // clamp loads; stores are guarded
  const __bf16* pa = (const __bf16*)A + (size_t)rowc * K;

  v8f acc0 = {}, acc1 = {}, acc2 = {}, acc3 = {};
  for (int k0 = 0; k0 < K; k0 += 32) {
    v16bf af = load_a_frag(pa, k0, g);
    const __bf16* pb0 = (const __bf16*)Bt + (size_t)(n0 + 0 * 16 + rlo) * K;
    const __bf16* pb1 = (const __bf16*)Bt + (size_t)(n0 + 1 * 16 + rlo) * K;
    const __bf16* pb2 = (const __bf16*)Bt + (size_t)(n0 + 2 * 16 + rlo) * K;
    const __bf16* pb3 = (const __bf16*)Bt + (size_t)(n0 + 3 * 16 + rlo) * K;
    acc0 = __builtin_amdgcn_wmma_f32_16x16x32_bf16(false, af, false,
             load_b_frag(pb0, k0, g), (short)0, acc0, false, false);
    acc1 = __builtin_amdgcn_wmma_f32_16x16x32_bf16(false, af, false,
             load_b_frag(pb1, k0, g), (short)0, acc1, false, false);
    acc2 = __builtin_amdgcn_wmma_f32_16x16x32_bf16(false, af, false,
             load_b_frag(pb2, k0, g), (short)0, acc2, false, false);
    acc3 = __builtin_amdgcn_wmma_f32_16x16x32_bf16(false, af, false,
             load_b_frag(pb3, k0, g), (short)0, acc3, false, false);
  }

  // C/D layout: vgpr r -> M = m0 + r + 8*g ; N = n0 + t*16 + (lane&15)
  v8f accs[4] = {acc0, acc1, acc2, acc3};
#pragma unroll
  for (int t = 0; t < 4; ++t) {
    int n = n0 + t * 16 + rlo;
    float bb = bias[n];
#pragma unroll
    for (int r = 0; r < 8; ++r) {
      int m = m0 + r + 8 * g;
      if (m < M) C[(size_t)m * Ncol + n] = accs[t][r] + bb;
    }
  }
}

// --------------------------- edge-phase kernels ----------------------------
// one wave per edge; edges [0,E) from lists, [E, E+N) are self loops

__global__ void k_edge_logits(const int* __restrict__ src, const int* __restrict__ dst,
                              int E, int N,
                              const float* __restrict__ xl, const float* __restrict__ xr,
                              const float* __restrict__ att,
                              float* __restrict__ abuf, unsigned* __restrict__ mkey) {
  int w = (int)((blockIdx.x * blockDim.x + threadIdx.x) >> 5);
  int lane = threadIdx.x & 31;
  if (w >= E + N) return;                      // wave-uniform
  int s, d;
  if (w < E) { s = src[w]; d = dst[w]; } else { s = w - E; d = s; }
  const float* pl = xl + (size_t)s * HC;
  const float* pr = xr + (size_t)d * HC;
  int c0 = lane * 4;
  float lg[NH];
#pragma unroll
  for (int h = 0; h < NH; ++h) {
    float4 vl = *(const float4*)(pl + h * HID + c0);
    float4 vr = *(const float4*)(pr + h * HID + c0);
    float4 va = *(const float4*)(att + h * HID + c0);
    float e0 = vl.x + vr.x; e0 = e0 > 0.f ? e0 : 0.2f * e0;
    float e1 = vl.y + vr.y; e1 = e1 > 0.f ? e1 : 0.2f * e1;
    float e2 = vl.z + vr.z; e2 = e2 > 0.f ? e2 : 0.2f * e2;
    float e3 = vl.w + vr.w; e3 = e3 > 0.f ? e3 : 0.2f * e3;
    float p = va.x * e0 + va.y * e1 + va.z * e2 + va.w * e3;
#pragma unroll
    for (int off = 16; off > 0; off >>= 1) p += __shfl_xor(p, off, 32);
    lg[h] = p;
  }
  if (lane == 0) {
#pragma unroll
    for (int h = 0; h < NH; ++h) {
      abuf[(size_t)w * NH + h] = lg[h];
      atomicMax(&mkey[(size_t)d * NH + h], fkey(lg[h]));
    }
  }
}

__global__ void k_edge_exp(const int* __restrict__ dst, int E, int N,
                           float* __restrict__ abuf, const unsigned* __restrict__ mkey,
                           float* __restrict__ denom) {
  long i = (long)blockIdx.x * blockDim.x + threadIdx.x;
  long tot = (long)(E + N) * NH;
  if (i >= tot) return;
  int e = (int)(i >> 2), h = (int)(i & 3);
  int d = (e < E) ? dst[e] : (e - E);
  float m = keyf(mkey[(size_t)d * NH + h]);
  float a = __expf(abuf[i] - m);
  abuf[i] = a;
  atomicAdd(&denom[(size_t)d * NH + h], a);
}

// acc[d,c] += (1/H) * sum_h a[e,h] * xl[s,h,c]   (head-mean folded in)
__global__ void k_edge_aggr(const int* __restrict__ src, const int* __restrict__ dst,
                            int E, int N,
                            const float* __restrict__ xl, const float* __restrict__ abuf,
                            const float* __restrict__ denom, float* __restrict__ hsum) {
  int w = (int)((blockIdx.x * blockDim.x + threadIdx.x) >> 5);
  int lane = threadIdx.x & 31;
  if (w >= E + N) return;
  int s, d;
  if (w < E) { s = src[w]; d = dst[w]; } else { s = w - E; d = s; }
  float a[NH];
#pragma unroll
  for (int h = 0; h < NH; ++h)
    a[h] = abuf[(size_t)w * NH + h] / denom[(size_t)d * NH + h];
  const float* pl = xl + (size_t)s * HC;
  float* po = hsum + (size_t)d * HID;
#pragma unroll
  for (int j = 0; j < 4; ++j) {
    int c = lane + 32 * j;
    float v = a[0] * pl[c] + a[1] * pl[HID + c] + a[2] * pl[2 * HID + c] +
              a[3] * pl[3 * HID + c];
    atomicAdd(po + c, 0.25f * v);
  }
}

// ------------------------------ GraphNorm ----------------------------------
// blockDim = 128 (thread = channel), block-strided over rows -> coalesced

__global__ void k_colsum(const float* __restrict__ x, float* __restrict__ sumbuf,
                         int N) {
  int c = threadIdx.x;
  float p = 0.f;
  for (int r = blockIdx.x; r < N; r += gridDim.x) p += x[(size_t)r * HID + c];
  atomicAdd(&sumbuf[c], p);
}

__global__ void k_colvar(const float* __restrict__ x, const float* __restrict__ sumbuf,
                         float* __restrict__ sqbuf, const float* __restrict__ ms,
                         int N) {
  int c = threadIdx.x;
  float mean = sumbuf[c] / (float)N;
  float sub = mean * ms[c];
  float p = 0.f;
  for (int r = blockIdx.x; r < N; r += gridDim.x) {
    float v = x[(size_t)r * HID + c] - sub;
    p += v * v;
  }
  atomicAdd(&sqbuf[c], p);
}

__global__ void k_apply(const float* __restrict__ hin, const float* __restrict__ hsum,
                        const float* __restrict__ sumbuf, const float* __restrict__ sqbuf,
                        const float* __restrict__ w, const float* __restrict__ b,
                        const float* __restrict__ ms, float* __restrict__ hout,
                        int N) {
  long i = (long)blockIdx.x * blockDim.x + threadIdx.x;
  if (i >= (long)N * HID) return;
  int c = (int)(i & (HID - 1));
  float invN = 1.f / (float)N;
  float mean = sumbuf[c] * invN;
  float var = sqbuf[c] * invN;
  float g = (hsum[i] - mean * ms[c]) * rsqrtf(var + 1e-5f) * w[c] + b[c];
  float gl = 0.5f * g * (1.f + erff(g * 0.70710678118654752f));  // exact gelu
  hout[i] = hin[i] + gl;
}

// ------------------------------ host driver --------------------------------

extern "C" void kernel_launch(void* const* d_in, const int* in_sizes, int n_in,
                              void* d_out, int out_size, void* d_ws, size_t ws_size,
                              hipStream_t stream) {
  (void)n_in; (void)out_size; (void)ws_size;
  const float* x     = (const float*)d_in[0];
  const int*   e_j   = (const int*)d_in[1];
  const int*   e_m   = (const int*)d_in[2];
  const float* W_in  = (const float*)d_in[3];
  const float* b_in  = (const float*)d_in[4];
  const float* Wl    = (const float*)d_in[5];
  const float* bl    = (const float*)d_in[6];
  const float* Wr    = (const float*)d_in[7];
  const float* br    = (const float*)d_in[8];
  const float* att   = (const float*)d_in[9];
  const float* bias  = (const float*)d_in[10];
  const float* gn_w  = (const float*)d_in[11];
  const float* gn_b  = (const float*)d_in[12];
  const float* gn_ms = (const float*)d_in[13];
  const float* W_out = (const float*)d_in[14];
  const float* b_out = (const float*)d_in[15];

  const int IN = 64, OUT = 64;
  const int N  = in_sizes[0] / IN;
  const int EJ = in_sizes[1] / 2;
  const int EM = in_sizes[2] / 2;
  const int Emax = EJ > EM ? EJ : EM;

  size_t off = 0;
  auto carve = [&](size_t bytes) -> void* {
    void* p = (char*)d_ws + off;
    off += (bytes + 255) & ~(size_t)255;
    return p;
  };
  float*          hA    = (float*)carve((size_t)N * HID * 4);
  float*          hB    = (float*)carve((size_t)N * HID * 4);
  float*          hsum  = (float*)carve((size_t)N * HID * 4);
  float*          xl    = (float*)carve((size_t)N * HC * 4);
  float*          xr    = (float*)carve((size_t)N * HC * 4);
  float*          abuf  = (float*)carve((size_t)(Emax + N) * NH * 4);
  unsigned*       mkey  = (unsigned*)carve((size_t)N * NH * 4);
  float*          denom = (float*)carve((size_t)N * NH * 4);
  float*          stat  = (float*)carve(2 * HID * 4);
  unsigned short* hbf   = (unsigned short*)carve((size_t)N * HID * 2);
  unsigned short* wtb   = (unsigned short*)carve((size_t)HC * HID * 2);

  auto cdiv = [](long a, long b) -> unsigned { return (unsigned)((a + b - 1) / b); };

  // -------- input projection: hA = x @ W_in + b_in --------
  long nx = (long)N * IN;
  k_tobf<<<cdiv(nx, 256), 256, 0, stream>>>(x, hbf, nx);
  k_wtrans<<<cdiv(IN * HID, 256), 256, 0, stream>>>(W_in, wtb, IN, HID);
  {
    dim3 g(cdiv(N, 128), HID / 64);
    k_gemm_bf16<<<g, 256, 0, stream>>>(hbf, wtb, b_in, hA, N, IN, HID);
  }

  float* hin = hA;
  float* hout = hB;
  long nh = (long)N * HID;
  for (int l = 0; l < 2; ++l) {
    k_tobf<<<cdiv(nh, 256), 256, 0, stream>>>(hin, hbf, nh);
    k_init_hsum<<<cdiv(nh, 256), 256, 0, stream>>>(
        hsum, bias + (size_t)(l * 2 + 0) * HID, bias + (size_t)(l * 2 + 1) * HID, nh);

    for (int r = 0; r < 2; ++r) {
      const int* src = r ? e_m : e_j;
      const int  E   = r ? EM  : EJ;
      const int* dst = src + E;
      size_t wo = (size_t)(l * 2 + r);

      k_wtrans<<<cdiv((long)HID * HC, 256), 256, 0, stream>>>(
          Wl + wo * HID * HC, wtb, HID, HC);
      dim3 gg(cdiv(N, 128), HC / 64);
      k_gemm_bf16<<<gg, 256, 0, stream>>>(hbf, wtb, bl + wo * HC, xl, N, HID, HC);
      k_wtrans<<<cdiv((long)HID * HC, 256), 256, 0, stream>>>(
          Wr + wo * HID * HC, wtb, HID, HC);
      k_gemm_bf16<<<gg, 256, 0, stream>>>(hbf, wtb, br + wo * HC, xr, N, HID, HC);

      long nm = (long)N * NH;
      k_zero<<<cdiv(nm, 256), 256, 0, stream>>>(mkey, nm);
      k_zero<<<cdiv(nm, 256), 256, 0, stream>>>((unsigned*)denom, nm);

      int tot = E + N;
      unsigned wblocks = cdiv(tot, 8);   // 8 waves / 256-thread block
      k_edge_logits<<<wblocks, 256, 0, stream>>>(src, dst, E, N, xl, xr,
                                                 att + wo * HC, abuf, mkey);
      k_edge_exp<<<cdiv((long)tot * NH, 256), 256, 0, stream>>>(dst, E, N, abuf,
                                                                mkey, denom);
      k_edge_aggr<<<wblocks, 256, 0, stream>>>(src, dst, E, N, xl, abuf, denom,
                                               hsum);
    }

    k_zero<<<1, 256, 0, stream>>>((unsigned*)stat, 2 * HID);
    k_colsum<<<512, HID, 0, stream>>>(hsum, stat, N);
    k_colvar<<<512, HID, 0, stream>>>(hsum, stat, stat + HID, gn_ms + (size_t)l * HID, N);
    k_apply<<<cdiv(nh, 256), 256, 0, stream>>>(
        hin, hsum, stat, stat + HID, gn_w + (size_t)l * HID, gn_b + (size_t)l * HID,
        gn_ms + (size_t)l * HID, hout, N);

    float* t = hin; hin = hout; hout = t;
  }

  // -------- output projection: d_out = h2 @ W_out + b_out --------
  k_tobf<<<cdiv(nh, 256), 256, 0, stream>>>(hin, hbf, nh);
  k_wtrans<<<cdiv(HID * OUT, 256), 256, 0, stream>>>(W_out, wtb, HID, OUT);
  {
    dim3 g(cdiv(N, 128), OUT / 64);
    k_gemm_bf16<<<g, 256, 0, stream>>>(hbf, wtb, b_out, (float*)d_out, N, HID, OUT);
  }
}
